// RNN_chaotic_17970143167382
// MI455X (gfx1250) — matrix operand order repaired and checked by reading, not compiled, not run on hardware
//
#include <hip/hip_runtime.h>
#include <math.h>

// Problem constants (from reference)
#define ISZ    256
#define HSZ    2048
#define OSZ    128
#define TSTEPS 8192
#define ALPHA  0.1f

// Persistent-kernel geometry
#define NWG   256           // workgroups (must all be co-resident)
#define RPW   8             // hidden rows per workgroup: 256*8 = 2048
#define BLK   256           // 8 wave32 waves; wave w owns row w of the slab

typedef __attribute__((ext_vector_type(16))) __bf16 v16bf;
typedef __attribute__((ext_vector_type(8)))  float  v8f;

// ---------------------------------------------------------------------------
// Kernel 0: init r0 = rate, clear grid barrier counter
// ---------------------------------------------------------------------------
__global__ void init_kernel(const float* __restrict__ rate,
                            float* __restrict__ r0,
                            unsigned* __restrict__ bar) {
    if (blockIdx.x == 0 && threadIdx.x == 0) *bar = 0u;
    for (int i = blockIdx.x * blockDim.x + threadIdx.x; i < HSZ;
         i += blockDim.x * gridDim.x) {
        r0[i] = rate[i];
    }
}

// ---------------------------------------------------------------------------
// Kernel 1: transpose input_sig (I, T) -> sigT (T, I) so each step reads a
// contiguous 1KB row (L2-friendly broadcast across all WGs).
// ---------------------------------------------------------------------------
__global__ __launch_bounds__(256) void transpose_sig(const float* __restrict__ in,
                                                     float* __restrict__ out) {
    __shared__ float tile[32][33];
    const int tx = threadIdx.x & 31;
    const int ty = threadIdx.x >> 5;            // 0..7
    const int tBase = blockIdx.x * 32;          // along T
    const int iBase = blockIdx.y * 32;          // along I
    for (int r = ty; r < 32; r += 8)
        tile[r][tx] = in[(size_t)(iBase + r) * TSTEPS + tBase + tx];
    __syncthreads();
    for (int r = ty; r < 32; r += 8)
        out[(size_t)(tBase + r) * ISZ + iBase + tx] = tile[tx][r];
}

// ---------------------------------------------------------------------------
// Kernel 2: persistent recurrence. Wh slab (bf16) + Wi slab (f32) pinned in
// LDS; r double-buffered in global (L2-resident); grid-wide split barrier per
// step using a monotonically increasing counter. Writes rates directly into
// d_out's rate_all region in (H, T) layout. Inner loop uses float4 / uint2
// LDS accesses (conflict-free, b128/b64 ds ops) with bf16 unpack via bit-ops.
// ---------------------------------------------------------------------------
__global__ __launch_bounds__(BLK) void rnn_persistent(
        const float* __restrict__ sigT,   // (T, I)
        const float* __restrict__ Wi,     // (H, I)
        const float* __restrict__ bi,     // (H)
        const float* __restrict__ Wh,     // (H, H)
        const float* __restrict__ bh,     // (H)
        float* __restrict__ r_a,          // (H) state buffer A (initialized)
        float* __restrict__ r_b,          // (H) state buffer B
        unsigned* __restrict__ bar,       // grid barrier counter (init 0)
        float* __restrict__ rate_out)     // (H, T) region of d_out
{
    __shared__ __bf16 sWh[RPW][HSZ];      // 32 KB (bf16 weights)
    __shared__ float  sWi[RPW][ISZ];      // 8 KB
    __shared__ float  sR[HSZ];            // 8 KB (current state)
    __shared__ float  sB[RPW];            // fused bi+bh

    const int tid  = threadIdx.x;
    const int wg   = blockIdx.x;
    const int row0 = wg * RPW;
    const int wave = tid >> 5;
    const int lane = tid & 31;
    const int rowg = row0 + wave;

    // One-time slab load into LDS
    for (int idx = tid; idx < RPW * HSZ; idx += BLK) {
        const int r = idx >> 11, k = idx & (HSZ - 1);
        sWh[r][k] = (__bf16)Wh[(size_t)(row0 + r) * HSZ + k];
    }
    for (int idx = tid; idx < RPW * ISZ; idx += BLK) {
        const int r = idx >> 8, k = idx & (ISZ - 1);
        sWi[r][k] = Wi[(size_t)(row0 + r) * ISZ + k];
    }
    if (tid < RPW) sB[tid] = bi[row0 + tid] + bh[row0 + tid];
    __syncthreads();

    const float* rcur = r_a;
    float*       rnxt = r_b;

    const uint2*  wh2 = (const uint2*)&sWh[wave][0];   // 2 bf16 pairs per elem
    const float4* wi4 = (const float4*)&sWi[wave][0];
    const float4* sR4c = (const float4*)sR;
    float4*       sR4  = (float4*)sR;

    for (int t = 0; t < TSTEPS; ++t) {
        // stage current state into LDS with b128 copies (8 KB, hot in L2)
        {
            const float4* rc4 = (const float4*)rcur;
            for (int i = tid; i < HSZ / 4; i += BLK) sR4[i] = rc4[i];
        }
        __syncthreads();

        const float* sig = sigT + (size_t)t * ISZ;
        if (t + 1 < TSTEPS)
            __builtin_prefetch(sig + ISZ, 0, 0);   // global_prefetch_b8

        float acc = 0.0f;

        // fused input projection: 2 x float4 per lane
        const float4* sig4 = (const float4*)sig;
#pragma unroll
        for (int kc = lane; kc < ISZ / 4; kc += 32) {
            const float4 s = sig4[kc];
            const float4 w = wi4[kc];
            acc = __builtin_fmaf(s.x, w.x, acc);
            acc = __builtin_fmaf(s.y, w.y, acc);
            acc = __builtin_fmaf(s.z, w.z, acc);
            acc = __builtin_fmaf(s.w, w.w, acc);
        }

        // recurrent mat-vec: 16 x (float4 state, uint2 bf16-weight) per lane
#pragma unroll 4
        for (int kc = lane; kc < HSZ / 4; kc += 32) {
            const float4 r = sR4c[kc];
            const uint2  w = wh2[kc];
            acc = __builtin_fmaf(r.x, __uint_as_float(w.x << 16),          acc);
            acc = __builtin_fmaf(r.y, __uint_as_float(w.x & 0xffff0000u),  acc);
            acc = __builtin_fmaf(r.z, __uint_as_float(w.y << 16),          acc);
            acc = __builtin_fmaf(r.w, __uint_as_float(w.y & 0xffff0000u),  acc);
        }

        acc += __shfl_down(acc, 16, 32);
        acc += __shfl_down(acc, 8, 32);
        acc += __shfl_down(acc, 4, 32);
        acc += __shfl_down(acc, 2, 32);
        acc += __shfl_down(acc, 1, 32);

        if (lane == 0) {
            const float rnew = (1.0f - ALPHA) * sR[rowg]
                             + ALPHA * tanhf(acc + sB[wave]);
            rnxt[rowg] = rnew;
            // consecutive t hit the same cacheline per row -> L2 write-combines
            rate_out[(size_t)rowg * TSTEPS + t] = rnew;
        }

        // ---- device-wide split barrier ----
        __syncthreads();
        __threadfence();                     // release writes to rnxt
        if (tid == 0) {
            atomicAdd(bar, 1u);
            const unsigned target = (unsigned)(t + 1) * (unsigned)NWG;
            while (__hip_atomic_load(bar, __ATOMIC_ACQUIRE,
                                     __HIP_MEMORY_SCOPE_AGENT) < target)
                __builtin_amdgcn_s_sleep(2);
        }
        __syncthreads();
        __threadfence();                     // acquire other WGs' writes

        const float* tmp = rnxt; rnxt = (float*)rcur; rcur = tmp;
    }
}

// ---------------------------------------------------------------------------
// Kernel 3: logits = rates @ Wo^T + bo via v_wmma_f32_16x16x32_bf16.
// A[t][h] = rate_all[h*T + t] (rate_all is (H,T)), B[h][o] = Wo[o*H + h].
// One 16x16 (t,o) tile per wave; 8 waves/block cover all 128 classes;
// grid = T/16 = 512 blocks. K = 2048, unrolled 2x32 per iteration so two
// chained WMMAs sit in the loop body.
// ---------------------------------------------------------------------------
__device__ __forceinline__ void load_frags(const float* __restrict__ rateall,
                                           const float* __restrict__ Wo,
                                           int k0, int m0, int n0,
                                           int l15, int hi,
                                           v16bf& a, v16bf& b) {
#pragma unroll
    for (int e = 0; e < 16; ++e) {
        // A fragment (16-bit A 16x32 ISA layout): rows in-lane
        const int ka = k0 + hi * 8 + e + ((e >= 8) ? 8 : 0);
        a[e] = (__bf16)rateall[(size_t)ka * TSTEPS + (m0 + l15)];
        // B fragment: rows striped across lanes, K in same order as A
        const int kb = k0 + hi * 8 + e + ((e >= 8) ? 8 : 0);
        b[e] = (__bf16)Wo[(size_t)(n0 + l15) * HSZ + kb];
    }
}

__global__ __launch_bounds__(256) void logits_wmma(
        const float* __restrict__ rateall,   // (H, T)
        const float* __restrict__ Wo,        // (O, H)
        const float* __restrict__ bo,        // (O)
        float* __restrict__ logits)          // (T, O)
{
    const int wave = threadIdx.x >> 5;
    const int lane = threadIdx.x & 31;
    const int m0   = blockIdx.x * 16;   // time tile
    const int n0   = wave * 16;         // class tile
    const int l15  = lane & 15;
    const int hi   = lane >> 4;         // 0 or 1: lane half

    v8f c = {};
    for (int k0 = 0; k0 < HSZ; k0 += 64) {
        v16bf a0, b0, a1, b1;
        load_frags(rateall, Wo, k0,      m0, n0, l15, hi, a0, b0);
        load_frags(rateall, Wo, k0 + 32, m0, n0, l15, hi, a1, b1);
        c = __builtin_amdgcn_wmma_f32_16x16x32_bf16(
                false, a0, false, b0, (short)0, c, false, false);
        c = __builtin_amdgcn_wmma_f32_16x16x32_bf16(
                false, a1, false, b1, (short)0, c, false, false);
    }

    const float bias = bo[n0 + l15];
#pragma unroll
    for (int j = 0; j < 8; ++j) {
        const int m = m0 + j + hi * 8;          // C/D: VGPR j -> M=j / M=j+8
        logits[(size_t)m * OSZ + (n0 + l15)] = c[j] + bias;
    }
}

// ---------------------------------------------------------------------------
// Kernel 4: log-softmax over the 128 classes of each timestep.
// output (O, T): out[o*T + t] = logits[t,o] - logsumexp_t
// ---------------------------------------------------------------------------
__global__ __launch_bounds__(128) void logsoftmax_kernel(
        const float* __restrict__ logits,    // (T, O)
        float* __restrict__ out)             // (O, T)
{
    __shared__ float sh[OSZ];
    const int t = blockIdx.x;
    const int o = threadIdx.x;
    const float x = logits[(size_t)t * OSZ + o];

    sh[o] = x; __syncthreads();
    for (int s = 64; s > 0; s >>= 1) {
        if (o < s) sh[o] = fmaxf(sh[o], sh[o + s]);
        __syncthreads();
    }
    const float mx = sh[0];
    __syncthreads();
    sh[o] = __expf(x - mx); __syncthreads();
    for (int s = 64; s > 0; s >>= 1) {
        if (o < s) sh[o] += sh[o + s];
        __syncthreads();
    }
    const float lse = __logf(sh[0]) + mx;
    out[(size_t)o * TSTEPS + t] = x - lse;
}

// ---------------------------------------------------------------------------
extern "C" void kernel_launch(void* const* d_in, const int* in_sizes, int n_in,
                              void* d_out, int out_size, void* d_ws, size_t ws_size,
                              hipStream_t stream) {
    const float* input_sig = (const float*)d_in[0]; // (256, 8192)
    const float* rate      = (const float*)d_in[1]; // (2048)
    const float* Wi        = (const float*)d_in[2]; // (2048, 256)
    const float* bi        = (const float*)d_in[3]; // (2048)
    const float* Wh        = (const float*)d_in[4]; // (2048, 2048)
    const float* bh        = (const float*)d_in[5]; // (2048)
    const float* Wo        = (const float*)d_in[6]; // (128, 2048)
    const float* bo        = (const float*)d_in[7]; // (128)

    float* out_logsm = (float*)d_out;                          // (128, 8192)
    float* out_rate  = (float*)d_out + (size_t)OSZ * TSTEPS;   // (2048, 8192)

    // workspace layout (~12.6 MB)
    float* sigT   = (float*)d_ws;                              // T*I
    float* logits = sigT + (size_t)TSTEPS * ISZ;               // T*O
    float* r_a    = logits + (size_t)TSTEPS * OSZ;             // H
    float* r_b    = r_a + HSZ;                                 // H
    unsigned* bar = (unsigned*)(r_b + HSZ);                    // 1

    init_kernel<<<8, 256, 0, stream>>>(rate, r_a, bar);

    dim3 tgrid(TSTEPS / 32, ISZ / 32);
    transpose_sig<<<tgrid, 256, 0, stream>>>(input_sig, sigT);

    rnn_persistent<<<NWG, BLK, 0, stream>>>(sigT, Wi, bi, Wh, bh,
                                            r_a, r_b, bar, out_rate);

    logits_wmma<<<TSTEPS / 16, 256, 0, stream>>>(out_rate, Wo, bo, logits);

    logsoftmax_kernel<<<TSTEPS, OSZ, 0, stream>>>(logits, out_logsm);
}
